// GATv2_raw_att_68968584839886
// MI455X (gfx1250) — matrix operand order repaired and checked by reading, not compiled, not run on hardware
//
#include <hip/hip_runtime.h>
#include <stdint.h>

#define N_NODES 40000
#define N_EDGES 80000
#define N_GRAPH 64
#define N_HEADS 4

typedef __attribute__((ext_vector_type(16))) __bf16 v16bf;
typedef __attribute__((ext_vector_type(8)))  __bf16 v8bf;
typedef __attribute__((ext_vector_type(8)))  float  v8f;

// ---------------- helpers ----------------
__device__ __forceinline__ unsigned short f32_to_bf16_rte(float f) {
  unsigned u = __float_as_uint(f);
  u += 0x7FFFu + ((u >> 16) & 1u);       // round-to-nearest-even
  return (unsigned short)(u >> 16);
}
// order-preserving float -> u32 key for atomicMax-based segment max
__device__ __forceinline__ unsigned fkey(float f) {
  unsigned u = __float_as_uint(f);
  return (u & 0x80000000u) ? ~u : (u | 0x80000000u);
}
__device__ __forceinline__ float fdekey(unsigned k) {
  unsigned u = (k & 0x80000000u) ? (k & 0x7FFFFFFFu) : ~k;
  return __uint_as_float(u);
}

// ---------------- generic zero ----------------
__global__ void zero_kernel(unsigned* __restrict__ p, size_t n) {
  size_t i = (size_t)blockIdx.x * blockDim.x + threadIdx.x;
  if (i < n) p[i] = 0u;
}

// ---------------- fp32 -> bf16 conversions ----------------
__global__ void convert_x0(const float* __restrict__ x, unsigned short* __restrict__ xbf, int Nn) {
  int i = blockIdx.x * blockDim.x + threadIdx.x;
  if (i >= Nn * 32) return;
  int n = i >> 5, c = i & 31;
  xbf[i] = (c < 4) ? f32_to_bf16_rte(x[n * 4 + c]) : (unsigned short)0;
}
__global__ void convert_w(const float* __restrict__ w, unsigned short* __restrict__ wbf,
                          int ci, int Kp, int ncols) {
  int i = blockIdx.x * blockDim.x + threadIdx.x;
  if (i >= Kp * ncols) return;
  int k = i / ncols, n = i % ncols;
  wbf[i] = (k < ci) ? f32_to_bf16_rte(w[(size_t)k * ncols + n]) : (unsigned short)0;
}

// ---------------- bf16 WMMA GEMM with async-LDS B staging ----------------
// C[M,N] = A[M,K] @ B[K,N].  A: bf16 row-major (K mult of 32); B: bf16 row-major
// (N mult of 64); C: f32.  Block = 8 waves; each wave owns a 32x64 tile
// (2 A-frags x 4 B-frags = 8 WMMA accumulators).  The 32x64 B k-slice is shared
// by all waves: staged once per block into LDS via global_load_async_to_lds_b128
// (double buffered, ASYNCcnt), rows padded to 144B to spread LDS banks.
#define BROW 72  // padded LDS row stride in bf16 elements (144 bytes)

__global__ __launch_bounds__(256) void gemm_bf16_wmma(
    const unsigned short* __restrict__ A,
    const unsigned short* __restrict__ B,
    float* __restrict__ C, int M, int K, int N) {
  __shared__ __align__(16) unsigned short Bs[2][32 * BROW];

  const int tid  = threadIdx.x;
  const int lane = tid & 31;
  const int wave = tid >> 5;
  const int n0   = blockIdx.x * 64;
  const int half = lane >> 4;
  const int l15  = lane & 15;

  const int mtiles = M >> 5;                        // 32-row tiles (M % 32 == 0)
  const int mt     = blockIdx.y * 8 + wave;
  const int mtc    = mt < mtiles ? mt : mtiles - 1; // clamp: keep barriers/EXEC uniform
  const int m0     = mtc * 32;

  // staging role: thread t copies one 16B chunk: row r = t>>3, cols (t&7)*8..+7
  const int srow = tid >> 3;
  const int scol = (tid & 7) * 8;

  v8f z = {0.f,0.f,0.f,0.f,0.f,0.f,0.f,0.f};
  v8f acc00 = z, acc01 = z, acc02 = z, acc03 = z;   // rows m0..m0+15
  v8f acc10 = z, acc11 = z, acc12 = z, acc13 = z;   // rows m0+16..m0+31

  const unsigned short* arow0 = A + (size_t)(m0 + l15) * K;
  const unsigned short* arow1 = arow0 + (size_t)16 * K;

  auto stage = [&](int buf, int k0) {
    unsigned ldso = (unsigned)(size_t)&Bs[buf][srow * BROW + scol];
    const unsigned short* gp = B + (size_t)(k0 + srow) * N + n0 + scol;
    asm volatile("global_load_async_to_lds_b128 %0, %1, off"
                 :: "v"(ldso), "v"((unsigned long long)(size_t)gp)
                 : "memory");
  };

  stage(0, 0);
  for (int k0 = 0; k0 < K; k0 += 32) {
    const int buf = (k0 >> 5) & 1;
    const bool more = (k0 + 32) < K;
    if (more) {
      stage(buf ^ 1, k0 + 32);                       // next slice in flight
      asm volatile("s_wait_asynccnt 1" ::: "memory"); // current slice landed
    } else {
      asm volatile("s_wait_asynccnt 0" ::: "memory");
    }
    __syncthreads();                                  // all waves' copies visible

    // A fragments (ISA 16-bit 16x32 layout) straight from global
    v8bf a0lo = *reinterpret_cast<const v8bf*>(arow0 + k0 + half * 8);
    v8bf a0hi = *reinterpret_cast<const v8bf*>(arow0 + k0 + 16 + half * 8);
    v8bf a1lo = *reinterpret_cast<const v8bf*>(arow1 + k0 + half * 8);
    v8bf a1hi = *reinterpret_cast<const v8bf*>(arow1 + k0 + 16 + half * 8);
    v16bf a0 = __builtin_shufflevector(a0lo, a0hi, 0,1,2,3,4,5,6,7,8,9,10,11,12,13,14,15);
    v16bf a1 = __builtin_shufflevector(a1lo, a1hi, 0,1,2,3,4,5,6,7,8,9,10,11,12,13,14,15);
    __builtin_prefetch(arow0 + k0 + 32, 0, 0);        // global_prefetch_b8
    __builtin_prefetch(arow1 + k0 + 32, 0, 0);

    // B fragments from LDS: lane holds column n0+l15(+16j), element s at K = table order (same as A)
    const __bf16* bb = reinterpret_cast<const __bf16*>(&Bs[buf][l15]);
    v16bf b0, b1, b2, b3;
#pragma unroll
    for (int s = 0; s < 16; s++) {
      const int kk = (s < 8) ? (half * 8 + s) : (16 + half * 8 + (s - 8));
      const __bf16* br = bb + kk * BROW;
      b0[s] = br[0]; b1[s] = br[16]; b2[s] = br[32]; b3[s] = br[48];
    }

    acc00 = __builtin_amdgcn_wmma_f32_16x16x32_bf16(false, a0, false, b0, (short)0, acc00, false, false);
    acc01 = __builtin_amdgcn_wmma_f32_16x16x32_bf16(false, a0, false, b1, (short)0, acc01, false, false);
    acc02 = __builtin_amdgcn_wmma_f32_16x16x32_bf16(false, a0, false, b2, (short)0, acc02, false, false);
    acc03 = __builtin_amdgcn_wmma_f32_16x16x32_bf16(false, a0, false, b3, (short)0, acc03, false, false);
    acc10 = __builtin_amdgcn_wmma_f32_16x16x32_bf16(false, a1, false, b0, (short)0, acc10, false, false);
    acc11 = __builtin_amdgcn_wmma_f32_16x16x32_bf16(false, a1, false, b1, (short)0, acc11, false, false);
    acc12 = __builtin_amdgcn_wmma_f32_16x16x32_bf16(false, a1, false, b2, (short)0, acc12, false, false);
    acc13 = __builtin_amdgcn_wmma_f32_16x16x32_bf16(false, a1, false, b3, (short)0, acc13, false, false);

    __syncthreads();                                  // protect buffer reuse
  }

  if (mt < mtiles) {
    // C/D layout: VGPR i -> row (i + 8*half), col (lane&15)
    float* c0 = C + (size_t)(m0 + 8 * half) * N + n0 + l15;
    float* c1 = C + (size_t)(m0 + 16 + 8 * half) * N + n0 + l15;
#pragma unroll
    for (int i = 0; i < 8; i++) {
      float* p0 = c0 + (size_t)i * N;
      float* p1 = c1 + (size_t)i * N;
      p0[0] = acc00[i]; p0[16] = acc01[i]; p0[32] = acc02[i]; p0[48] = acc03[i];
      p1[0] = acc10[i]; p1[16] = acc11[i]; p1[32] = acc12[i]; p1[48] = acc13[i];
    }
  }
}

// ---------------- edge pass 1: alpha + segment max ----------------
__global__ void edge_alpha(const float* __restrict__ xl, const float* __restrict__ xr,
                           const float* __restrict__ ef, const float* __restrict__ we,
                           const float* __restrict__ att,
                           const int* __restrict__ src, const int* __restrict__ dst,
                           float* __restrict__ alpha, unsigned* __restrict__ amax,
                           int E, int co) {
  int t = blockIdx.x * blockDim.x + threadIdx.x;
  if (t >= E * N_HEADS) return;
  int e = t >> 2, h = t & 3;
  int HC = N_HEADS * co;
  const float* fe = ef + (size_t)e * 6;
  float f0 = fe[0], f1 = fe[1], f2 = fe[2], f3 = fe[3], f4 = fe[4], f5 = fe[5];
  const float* pl = xl + (size_t)src[e] * HC + h * co;
  const float* pr = xr + (size_t)dst[e] * HC + h * co;
  const float* pw = we + h * co;           // we: 6 x HC row-major
  const float* pa = att + h * co;
  float s = 0.f;
  for (int c = 0; c < co; c++) {
    float ee = f0 * pw[c] + f1 * pw[HC + c] + f2 * pw[2 * HC + c] +
               f3 * pw[3 * HC + c] + f4 * pw[4 * HC + c] + f5 * pw[5 * HC + c];
    float mm = pl[c] + pr[c] + ee;
    mm = mm > 0.f ? mm : 0.2f * mm;        // leaky_relu(0.2)
    s += mm * pa[c];
  }
  alpha[t] = s;
  atomicMax(&amax[(size_t)dst[e] * N_HEADS + h], fkey(s));
}

// ---------------- edge pass 2: exp + segment sum ----------------
__global__ void edge_exp(const float* __restrict__ alpha, const unsigned* __restrict__ amax,
                         const int* __restrict__ dst, float* __restrict__ exb,
                         float* __restrict__ den, int E) {
  int t = blockIdx.x * blockDim.x + threadIdx.x;
  if (t >= E * N_HEADS) return;
  int e = t >> 2, h = t & 3;
  int d = dst[e];
  float v = __expf(alpha[t] - fdekey(amax[(size_t)d * N_HEADS + h]));
  exb[t] = v;
  atomicAdd(&den[(size_t)d * N_HEADS + h], v);
}

// ---------------- edge pass 3: weighted scatter (mean over heads folded in) ----------------
__global__ void edge_aggregate(const float* __restrict__ xl, const float* __restrict__ exb,
                               const float* __restrict__ den,
                               const int* __restrict__ src, const int* __restrict__ dst,
                               float* __restrict__ outb, int E, int co) {
  int t = blockIdx.x * blockDim.x + threadIdx.x;
  if (t >= E * N_HEADS) return;
  int e = t >> 2, h = t & 3;
  int d = dst[e];
  float a = exb[t] / (den[(size_t)d * N_HEADS + h] + 1e-16f) * 0.25f;
  const float* pl = xl + (size_t)src[e] * (N_HEADS * co) + h * co;
  float* po = outb + (size_t)d * co;
  for (int c = 0; c < co; c++) atomicAdd(&po[c], a * pl[c]);
}

// ---------------- bias + BN statistics (LDS tree reduce per channel) ----------------
__global__ __launch_bounds__(256) void finalize_bnstats(
    const float* __restrict__ outb, const float* __restrict__ bias,
    float* __restrict__ xcur, float* __restrict__ bnsum, float* __restrict__ bnsq,
    int Nn, int co) {
  int c = blockIdx.y;
  int n = blockIdx.x * 256 + threadIdx.x;
  __shared__ float s1[256], s2[256];
  float v = 0.f, v2 = 0.f;
  if (n < Nn) {
    v = outb[(size_t)n * co + c] + bias[c];
    xcur[(size_t)n * co + c] = v;
    v2 = v * v;
  }
  int tid = threadIdx.x;
  s1[tid] = v; s2[tid] = v2;
  __syncthreads();
  for (int off = 128; off > 0; off >>= 1) {
    if (tid < off) { s1[tid] += s1[tid + off]; s2[tid] += s2[tid + off]; }
    __syncthreads();
  }
  if (tid == 0) { atomicAdd(&bnsum[c], s1[0]); atomicAdd(&bnsq[c], s2[0]); }
}

// ---------------- BN apply + ELU + bf16 conversion for next GEMM ----------------
__global__ void bn_elu_cvt(const float* __restrict__ bnsum, const float* __restrict__ bnsq,
                           const float* __restrict__ g, const float* __restrict__ b,
                           float* __restrict__ xcur, unsigned short* __restrict__ xbf,
                           int Nn, int co) {
  size_t i = (size_t)blockIdx.x * blockDim.x + threadIdx.x;
  if (i >= (size_t)Nn * co) return;
  int c = (int)(i % co);
  float inv_n = 1.0f / (float)Nn;
  float mean = bnsum[c] * inv_n;
  float var  = bnsq[c] * inv_n - mean * mean;
  float v = xcur[i];
  v = g[c] * (v - mean) * rsqrtf(var + 1e-5f) + b[c];
  v = v > 0.f ? v : __expf(v) - 1.f;       // ELU
  xcur[i] = v;
  xbf[i]  = f32_to_bf16_rte(v);            // co is a multiple of 32 -> next-layer K
}

// ---------------- gate MLP + graph segment max ----------------
__global__ void gate_kernel(const float* __restrict__ x, const float* __restrict__ w1,
                            const float* __restrict__ b1, const float* __restrict__ w2,
                            const float* __restrict__ b2, const int* __restrict__ batch,
                            float* __restrict__ gate, unsigned* __restrict__ gmax, int Nn) {
  int n = blockIdx.x * blockDim.x + threadIdx.x;
  if (n >= Nn) return;
  const float* xr = x + (size_t)n * 128;
  float g = b2[0];
  for (int j = 0; j < 64; j++) {
    float s = b1[j];
    for (int k = 0; k < 128; k++) s += xr[k] * w1[k * 64 + j];
    s = s > 0.f ? s : 0.f;
    g += s * w2[j];
  }
  gate[n] = g;
  atomicMax(&gmax[batch[n]], fkey(g));
}

__global__ void pool_weight(const float* __restrict__ gate, const unsigned* __restrict__ gmax,
                            const int* __restrict__ batch, float* __restrict__ gex,
                            float* __restrict__ gden, int Nn) {
  int n = blockIdx.x * blockDim.x + threadIdx.x;
  if (n >= Nn) return;
  int b = batch[n];
  float v = __expf(gate[n] - fdekey(gmax[b]));
  gex[n] = v;
  atomicAdd(&gden[b], v);
}

__global__ void pool_sum(const float* __restrict__ xcur, const float* __restrict__ gex,
                         const float* __restrict__ gden, const int* __restrict__ batch,
                         float* __restrict__ hpool, int Nn) {
  size_t i = (size_t)blockIdx.x * blockDim.x + threadIdx.x;
  if (i >= (size_t)Nn * 128) return;
  int n = (int)(i >> 7), c = (int)(i & 127);
  int b = batch[n];
  float w = gex[n] / (gden[b] + 1e-16f);
  atomicAdd(&hpool[b * 128 + c], w * xcur[i]);
}

// ---------------- FC head: 64x128 MLP + BN + log_softmax in one block ----------------
__global__ __launch_bounds__(256) void fc_head(
    const float* __restrict__ hpool,
    const float* w0, const float* b0, const float* g0, const float* bb0,
    const float* w1, const float* b1, const float* g1, const float* bb1,
    const float* w2, const float* b2, const float* g2, const float* bb2,
    const float* w3, const float* b3,
    float* __restrict__ out) {
  __shared__ float cur[64 * 128];
  __shared__ float nxt[64 * 64];
  __shared__ float mu[64], rv[64];
  int tid = threadIdx.x;
  for (int i = tid; i < 64 * 128; i += 256) cur[i] = hpool[i];
  __syncthreads();
  const float* Ws[4] = {w0, w1, w2, w3};
  const float* Bs4[4] = {b0, b1, b2, b3};
  const float* Gs[3] = {g0, g1, g2};
  const float* Bb[3] = {bb0, bb1, bb2};
  const int dims[5] = {128, 64, 32, 16, 2};
  for (int l = 0; l < 4; l++) {
    int fi = dims[l], fo = dims[l + 1];
    for (int i = tid; i < 64 * fo; i += 256) {
      int gg = i / fo, j = i % fo;
      float s = Bs4[l][j];
      const float* row = &cur[gg * fi];
      for (int k = 0; k < fi; k++) s += row[k] * Ws[l][k * fo + j];
      nxt[i] = s;
    }
    __syncthreads();
    if (l < 3) {
      if (tid < fo) {
        float m = 0.f;
        for (int gg = 0; gg < 64; gg++) m += nxt[gg * fo + tid];
        m *= (1.0f / 64.0f);
        float v = 0.f;
        for (int gg = 0; gg < 64; gg++) { float d = nxt[gg * fo + tid] - m; v += d * d; }
        v *= (1.0f / 64.0f);
        mu[tid] = m; rv[tid] = rsqrtf(v + 1e-5f);
      }
      __syncthreads();
      for (int i = tid; i < 64 * fo; i += 256) {
        int j = i % fo;
        float v = Gs[l][j] * (nxt[i] - mu[j]) * rv[j] + Bb[l][j];
        cur[i] = v > 0.f ? v : __expf(v) - 1.f;
      }
    } else {
      for (int i = tid; i < 64 * fo; i += 256) cur[i] = nxt[i];
    }
    __syncthreads();
  }
  if (tid < 64) {
    float a = cur[tid * 2], b = cur[tid * 2 + 1];
    float m = fmaxf(a, b);
    float lse = m + __logf(__expf(a - m) + __expf(b - m));
    out[tid * 2] = a - lse;
    out[tid * 2 + 1] = b - lse;
  }
}

// ---------------- host launcher ----------------
extern "C" void kernel_launch(void* const* d_in, const int* in_sizes, int n_in,
                              void* d_out, int out_size, void* d_ws, size_t ws_size,
                              hipStream_t stream) {
  (void)in_sizes; (void)n_in; (void)out_size; (void)ws_size;
  const int N = N_NODES, E = N_EDGES, H = N_HEADS;
  const int DIMS[8] = {4, 32, 64, 128, 256, 512, 256, 128};

  // --- input pointers (top-level dict insertion order; params tree-flattened alphabetically) ---
  const float* x_in  = (const float*)d_in[0];
  const int*   eidx  = (const int*)d_in[1];
  const float* efeat = (const float*)d_in[2];
  const int*   batch = (const int*)d_in[3];
  const int* srcv = eidx;        // edge_index row 0
  const int* dstv = eidx + E;    // edge_index row 1

  // convs[7], sorted keys: att, bias, bn_b, bn_g, we, wl, wr
  const float *c_att[7], *c_bias[7], *c_bnb[7], *c_bng[7], *c_we[7], *c_wl[7], *c_wr[7];
  for (int l = 0; l < 7; l++) {
    void* const* p = d_in + 4 + l * 7;
    c_att[l] = (const float*)p[0]; c_bias[l] = (const float*)p[1];
    c_bnb[l] = (const float*)p[2]; c_bng[l] = (const float*)p[3];
    c_we[l]  = (const float*)p[4]; c_wl[l]  = (const float*)p[5];
    c_wr[l]  = (const float*)p[6];
  }
  // fcs[0..2]: b, bn_b, bn_g, w ; fcs[3]: b, w
  const float *f_b[4], *f_bnb[3], *f_bng[3], *f_w[4];
  for (int i = 0; i < 3; i++) {
    void* const* p = d_in + 53 + i * 4;
    f_b[i] = (const float*)p[0]; f_bnb[i] = (const float*)p[1];
    f_bng[i] = (const float*)p[2]; f_w[i] = (const float*)p[3];
  }
  f_b[3] = (const float*)d_in[65]; f_w[3] = (const float*)d_in[66];
  // gate: b1, b2, w1, w2
  const float* g_b1 = (const float*)d_in[67];
  const float* g_b2 = (const float*)d_in[68];
  const float* g_w1 = (const float*)d_in[69];
  const float* g_w2 = (const float*)d_in[70];

  // --- workspace carving (256B aligned) ---
  char* wsp = (char*)d_ws;
  auto carve = [&](size_t bytes) -> void* {
    void* p = (void*)wsp;
    wsp += (bytes + 255) & ~(size_t)255;
    return p;
  };
  unsigned short* xbf  = (unsigned short*)carve((size_t)N * 512 * 2);
  float*          xcur = (float*)carve((size_t)N * 512 * 4);
  unsigned short* wbfL = (unsigned short*)carve((size_t)512 * 2048 * 2);
  unsigned short* wbfR = (unsigned short*)carve((size_t)512 * 2048 * 2);
  float*          xl   = (float*)carve((size_t)N * 2048 * 4);
  float*          xr   = (float*)carve((size_t)N * 2048 * 4);
  float*          alph = (float*)carve((size_t)E * H * 4);
  float*          exb  = (float*)carve((size_t)E * H * 4);
  unsigned*       amax = (unsigned*)carve((size_t)N * H * 4);
  float*          den  = (float*)carve((size_t)N * H * 4);
  float*          outb = (float*)carve((size_t)N * 512 * 4);
  float*          bnsum = (float*)carve(512 * 4);
  float*          bnsq  = (float*)carve(512 * 4);
  float*          gateb = (float*)carve((size_t)N * 4);
  float*          gexb  = (float*)carve((size_t)N * 4);
  unsigned*       gmaxb = (unsigned*)carve(64 * 4);
  float*          gdenb = (float*)carve(64 * 4);
  float*          hpool = (float*)carve(64 * 128 * 4);

  auto blocks = [](size_t n) { return dim3((unsigned)((n + 255) / 256)); };

  // layer-0 input to padded bf16
  convert_x0<<<blocks((size_t)N * 32), 256, 0, stream>>>(x_in, xbf, N);

  for (int l = 0; l < 7; l++) {
    int ci = DIMS[l], co = DIMS[l + 1];
    int Kp = ci < 32 ? 32 : ci;
    int HC = H * co;

    convert_w<<<blocks((size_t)Kp * HC), 256, 0, stream>>>(c_wl[l], wbfL, ci, Kp, HC);
    convert_w<<<blocks((size_t)Kp * HC), 256, 0, stream>>>(c_wr[l], wbfR, ci, Kp, HC);

    dim3 ggrid(HC / 64, (N / 32 + 7) / 8);
    gemm_bf16_wmma<<<ggrid, 256, 0, stream>>>(xbf, wbfL, xl, N, Kp, HC);
    gemm_bf16_wmma<<<ggrid, 256, 0, stream>>>(xbf, wbfR, xr, N, Kp, HC);

    zero_kernel<<<blocks((size_t)N * H), 256, 0, stream>>>(amax, (size_t)N * H);
    zero_kernel<<<blocks((size_t)N * H), 256, 0, stream>>>((unsigned*)den, (size_t)N * H);
    zero_kernel<<<blocks((size_t)N * co), 256, 0, stream>>>((unsigned*)outb, (size_t)N * co);
    zero_kernel<<<blocks((size_t)co * 2), 256, 0, stream>>>((unsigned*)bnsum, (size_t)co);
    zero_kernel<<<blocks((size_t)co * 2), 256, 0, stream>>>((unsigned*)bnsq, (size_t)co);

    edge_alpha<<<blocks((size_t)E * H), 256, 0, stream>>>(xl, xr, efeat, c_we[l], c_att[l],
                                                          srcv, dstv, alph, amax, E, co);
    edge_exp<<<blocks((size_t)E * H), 256, 0, stream>>>(alph, amax, dstv, exb, den, E);
    edge_aggregate<<<blocks((size_t)E * H), 256, 0, stream>>>(xl, exb, den, srcv, dstv, outb, E, co);

    dim3 fgrid((N + 255) / 256, co);
    finalize_bnstats<<<fgrid, 256, 0, stream>>>(outb, c_bias[l], xcur, bnsum, bnsq, N, co);
    bn_elu_cvt<<<blocks((size_t)N * co), 256, 0, stream>>>(bnsum, bnsq, c_bng[l], c_bnb[l],
                                                           xcur, xbf, N, co);
  }

  // gate + softmax pooling over graphs
  zero_kernel<<<1, 256, 0, stream>>>(gmaxb, 64);
  zero_kernel<<<1, 256, 0, stream>>>((unsigned*)gdenb, 64);
  zero_kernel<<<blocks(64 * 128), 256, 0, stream>>>((unsigned*)hpool, 64 * 128);
  gate_kernel<<<blocks((size_t)N), 256, 0, stream>>>(xcur, g_w1, g_b1, g_w2, g_b2,
                                                     batch, gateb, gmaxb, N);
  pool_weight<<<blocks((size_t)N), 256, 0, stream>>>(gateb, gmaxb, batch, gexb, gdenb, N);
  pool_sum<<<blocks((size_t)N * 128), 256, 0, stream>>>(xcur, gexb, gdenb, batch, hpool, N);

  fc_head<<<1, 256, 0, stream>>>(hpool,
                                 f_w[0], f_b[0], f_bng[0], f_bnb[0],
                                 f_w[1], f_b[1], f_bng[1], f_bnb[1],
                                 f_w[2], f_b[2], f_bng[2], f_bnb[2],
                                 f_w[3], f_b[3],
                                 (float*)d_out);
}
